// FiBiNETModel_15745350107449
// MI455X (gfx1250) — matrix operand order repaired and checked by reading, not compiled, not run on hardware
//
#include <hip/hip_runtime.h>

// ---------------- problem constants ----------------
#define B_SZ 4096
#define F_SZ 24
#define V_SZ 1000
#define D_SZ 32
#define RED  8
#define FD   768      // F*D
#define COMB 1536     // 2*F*D
#define H0N  1024
#define H1N  512

typedef unsigned short u16;
typedef __attribute__((ext_vector_type(16))) __bf16 v16bf;
typedef __attribute__((ext_vector_type(8)))  float  v8f;
typedef __attribute__((ext_vector_type(4)))  unsigned int uv4;
typedef __attribute__((address_space(3))) u16 lds_u16;

// ---------------- helpers ----------------
__device__ __forceinline__ u16 f2bf(float f) {
  unsigned int u = __float_as_uint(f);
  unsigned int r = u + 0x7FFFu + ((u >> 16) & 1u);   // round-to-nearest-even
  return (u16)(r >> 16);
}
__device__ __forceinline__ float bf2f(u16 h) {
  return __uint_as_float(((unsigned int)h) << 16);
}

// CDNA5 async global->LDS copy (ASYNCcnt path). The LDS destination is passed
// as an addrspace(3) pointer: its 32-bit value is the LDS byte offset the
// VDST operand expects, and passing the real pointer captures the shared
// object so LLVM cannot fold away the subsequent LDS reads.
__device__ __forceinline__ void async_copy_b128(u16* lds_p, const void* g) {
  lds_u16* lp = (lds_u16*)lds_p;                     // generic -> LDS addrspace
  asm volatile("global_load_async_to_lds_b128 %0, %1, off"
               :: "v"(lp), "v"(g) : "memory");
}
__device__ __forceinline__ void wait_async() {
#if __has_builtin(__builtin_amdgcn_s_wait_asynccnt)
  __builtin_amdgcn_s_wait_asynccnt(0);
#else
  asm volatile("s_wait_asynccnt 0x0" ::: "memory");
#endif
}

// ---------------- repack kernels ----------------
// Wbig[r=i*32+k][c=j*32+l] = bil_W[i][j][k][l]  -> bf16 row-major [768][768]
__global__ void k_repack_bilW(const float* __restrict__ W, u16* __restrict__ out) {
  int idx = blockIdx.x * 256 + threadIdx.x;          // 768*768 exact
  int r = idx / FD, c = idx % FD;
  int i = r >> 5, k = r & 31, j = c >> 5, l = c & 31;
  out[idx] = f2bf(W[(((size_t)(i * F_SZ + j) * D_SZ + k) * D_SZ) + l]);
}

// in [K][N] f32 row-major -> out [N][K] bf16 row-major
__global__ void k_transpose_bf16(const float* __restrict__ in, u16* __restrict__ out,
                                 int K, int N) {
  int idx = blockIdx.x * 256 + threadIdx.x;
  if (idx >= N * K) return;
  int n = idx / K, k = idx % K;
  out[idx] = f2bf(in[(size_t)k * N + n]);
}

// ---------------- embedding + linear + SENet ----------------
__global__ void k_embed_senet(const int* __restrict__ x,
                              const float* __restrict__ et,
                              const float* __restrict__ lt,
                              const float* __restrict__ sw1,
                              const float* __restrict__ sw2,
                              u16* __restrict__ E2,
                              float* __restrict__ lin) {
  __shared__ int   xs[F_SZ];
  __shared__ float se[FD];
  __shared__ float sZ[F_SZ];
  __shared__ float sT[RED];
  __shared__ float sA[F_SZ];
  __shared__ float slin[F_SZ];
  const int b = blockIdx.x, tid = threadIdx.x;

  if (tid < F_SZ) {
    int xi = x[b * F_SZ + tid];
    xs[tid] = xi;
    slin[tid] = lt[tid * V_SZ + xi];
  }
  __syncthreads();
  for (int r = tid; r < FD; r += 256) {
    int f = r >> 5, d = r & 31;
    float v = et[((size_t)f * V_SZ + xs[f]) * D_SZ + d];
    se[r] = v;
    E2[(size_t)b * FD + r] = f2bf(v);
  }
  __syncthreads();
  if (tid < F_SZ) {                                  // squeeze: mean over D
    float s = 0.f;
    for (int d = 0; d < D_SZ; ++d) s += se[tid * D_SZ + d];
    sZ[tid] = s * (1.0f / D_SZ);
  }
  __syncthreads();
  if (tid < RED) {                                   // relu(Z @ w1^T)
    float s = 0.f;
    for (int f = 0; f < F_SZ; ++f) s += sw1[tid * F_SZ + f] * sZ[f];
    sT[tid] = s > 0.f ? s : 0.f;
  }
  __syncthreads();
  if (tid < F_SZ) {                                  // sigmoid(T @ w2^T)
    float s = 0.f;
    for (int r = 0; r < RED; ++r) s += sw2[tid * RED + r] * sT[r];
    sA[tid] = 1.0f / (1.0f + __expf(-s));
  }
  if (tid == 0) {
    float s = 0.f;
    for (int f = 0; f < F_SZ; ++f) s += slin[f];
    lin[b] = s;
  }
  __syncthreads();
  for (int r = tid; r < FD; r += 256) {
    int f = r >> 5;
    E2[(size_t)(B_SZ + b) * FD + r] = f2bf(se[r] * sA[f]);
  }
}

// ---------------- bf16 WMMA GEMM: C[M,N] = A[M,K] * BT[N,K]^T ----------------
// Block tile 128(M) x 128(N), 8 waves (4 along M x 2 along N), each wave a
// 32x64 region = 2x4 of 16x16 v_wmma_f32_16x16x32_bf16 tiles.
// Global->LDS staging via CDNA5 async-to-LDS DMA with double buffering.
// M%128==0, N%128==0, K%64==0 guaranteed by the problem sizes.
#define APAD 72                      // padded LDS row stride in u16 (144 B)

struct GemmLds {
  u16 A[2][128 * APAD];
  u16 B[2][128 * APAD];
};

__global__ __launch_bounds__(256) void k_gemm_bf16(
    const u16* __restrict__ A, const u16* __restrict__ BT, float* __restrict__ C,
    int M, int N, int K) {
  __shared__ GemmLds lds;

  const int tid   = threadIdx.x;
  const int lane  = tid & 31;
  const int wave  = tid >> 5;
  const int waveM = wave & 3;        // 4 waves along M (32 rows each)
  const int waveN = wave >> 2;       // 2 waves along N (64 cols each)
  const int blockM = blockIdx.y * 128;
  const int blockN = blockIdx.x * 128;
  const int grp = lane >> 4;         // lane half (ISA fragment layout)
  const int hl  = lane & 15;

  v8f acc[2][4] = {};

  const int row = tid >> 3;          // 0..31 base row for this thread's copies
  const int cb  = tid & 7;           // 16-byte column block within 64-wide tile

  // issue async copies of one 128x64 tile (A or B) into LDS buffer dstBase
  auto issue_tile = [&](u16* dstBase, const u16* __restrict__ src,
                        int gRow0, int k0) {
#pragma unroll
    for (int q = 0; q < 4; ++q) {
      int r = row + q * 32;
      async_copy_b128(dstBase + r * APAD + cb * 8,
                      src + (size_t)(gRow0 + r) * K + k0 + cb * 8);
    }
  };

  const int nT = K / 64;
  issue_tile(lds.A[0], A,  blockM, 0);
  issue_tile(lds.B[0], BT, blockN, 0);
  wait_async();
  __syncthreads();

  for (int t = 0; t < nT; ++t) {
    const int buf = t & 1;
    if (t + 1 < nT) {                // prefetch next tile while we compute
      issue_tile(lds.A[buf ^ 1], A,  blockM, (t + 1) * 64);
      issue_tile(lds.B[buf ^ 1], BT, blockN, (t + 1) * 64);
    }

#pragma unroll
    for (int kk = 0; kk < 64; kk += 32) {
      v16bf af[2], bfv[4];
#pragma unroll
      for (int sm = 0; sm < 2; ++sm) {
        // A 16x32 bf16 fragment: lane half selects K offset (+8 / +24)
        const u16* p = &lds.A[buf][(waveM * 32 + sm * 16 + hl) * APAD + kk + 8 * grp];
        *((uv4*)&af[sm])     = *(const uv4*)(p);
        *((uv4*)&af[sm] + 1) = *(const uv4*)(p + 16);
      }
#pragma unroll
      for (int sn = 0; sn < 4; ++sn) {
        // B 32x16 bf16 fragment: lane half selects K offset (+8 / +24)
        const u16* p = &lds.B[buf][(waveN * 64 + sn * 16 + hl) * APAD + kk + 8 * grp];
        *((uv4*)&bfv[sn])     = *(const uv4*)(p);
        *((uv4*)&bfv[sn] + 1) = *(const uv4*)(p + 16);
      }
#pragma unroll
      for (int sm = 0; sm < 2; ++sm)
#pragma unroll
        for (int sn = 0; sn < 4; ++sn)
          acc[sm][sn] = __builtin_amdgcn_wmma_f32_16x16x32_bf16(
              false, af[sm], false, bfv[sn], (short)0, acc[sm][sn], false, false);
    }

    wait_async();                    // next tile's DMA done (overlapped w/ WMMA)
    __syncthreads();                 // all waves done reading `buf`
  }

  // ---- epilogue: C/D layout  m = r + 8*grp, n = lane&15 ----
#pragma unroll
  for (int sm = 0; sm < 2; ++sm)
#pragma unroll
    for (int sn = 0; sn < 4; ++sn) {
      float* p = C + (size_t)(blockM + waveM * 32 + sm * 16 + grp * 8) * N
                   + blockN + waveN * 64 + sn * 16 + hl;
#pragma unroll
      for (int r = 0; r < 8; ++r)
        p[(size_t)r * N] = acc[sm][sn][r];
    }
}

// ---------------- diag correction + e*s + concat -> combined bf16 ----------------
__global__ void k_combine(const float* __restrict__ S, const u16* __restrict__ E2,
                          const float* __restrict__ W, u16* __restrict__ comb) {
  __shared__ float ep[FD], eq[FD];
  const int b = blockIdx.x, tid = threadIdx.x;
  for (int r = tid; r < FD; r += 256) {
    ep[r] = bf2f(E2[(size_t)b * FD + r]);
    eq[r] = bf2f(E2[(size_t)(B_SZ + b) * FD + r]);
  }
  __syncthreads();
  for (int r = tid; r < FD; r += 256) {
    int i = r >> 5, k = r & 31;
    const float* wd = W + ((size_t)(i * (F_SZ + 1)) * D_SZ + k) * D_SZ; // W[i][i][k][:]
    const float* e1 = ep + i * D_SZ;
    const float* e2 = eq + i * D_SZ;
    float dp = 0.f, dq = 0.f;
#pragma unroll
    for (int l = 0; l < D_SZ; ++l) { float w = wd[l]; dp += w * e1[l]; dq += w * e2[l]; }
    float p = ep[r] * (S[(size_t)b * FD + r] - dp);
    float q = eq[r] * (S[(size_t)(B_SZ + b) * FD + r] - dq);
    comb[(size_t)b * COMB + r]      = f2bf(p);
    comb[(size_t)b * COMB + FD + r] = f2bf(q);
  }
}

// ---------------- per-column batch stats (mean, 1/sqrt(var+eps)) ----------------
__global__ void k_colstats(const float* __restrict__ H, float* __restrict__ mean,
                           float* __restrict__ istd, int N) {
  __shared__ float ssum[4][64], ssq[4][64];
  const int tx = threadIdx.x & 63, ty = threadIdx.x >> 6;
  const int c = blockIdx.x * 64 + tx;
  float s = 0.f, s2 = 0.f;
  for (int b = ty; b < B_SZ; b += 4) {
    float v = H[(size_t)b * N + c];
    s += v; s2 += v * v;
  }
  ssum[ty][tx] = s; ssq[ty][tx] = s2;
  __syncthreads();
  if (ty == 0) {
    s  = ssum[0][tx] + ssum[1][tx] + ssum[2][tx] + ssum[3][tx];
    s2 = ssq[0][tx]  + ssq[1][tx]  + ssq[2][tx]  + ssq[3][tx];
    float m   = s * (1.0f / B_SZ);
    float var = s2 * (1.0f / B_SZ) - m * m;
    mean[c] = m;
    istd[c] = rsqrtf(var + 1e-5f);
  }
}

// ---------------- BN + ReLU -> bf16 ----------------
__global__ void k_bnrelu_bf16(const float* __restrict__ H, const float* __restrict__ mean,
                              const float* __restrict__ istd, const float* __restrict__ g,
                              const float* __restrict__ be, u16* __restrict__ out,
                              int N, int total) {
  int idx = blockIdx.x * 256 + threadIdx.x;
  if (idx >= total) return;
  int n = idx % N;
  float v = g[n] * (H[idx] - mean[n]) * istd[n] + be[n];
  out[idx] = f2bf(v > 0.f ? v : 0.f);
}

// ---------------- final: BN + ReLU + dot(w2) + lin + bias + sigmoid ----------------
__global__ void k_final(const float* __restrict__ H1, const float* __restrict__ mean,
                        const float* __restrict__ istd, const float* __restrict__ g,
                        const float* __restrict__ be, const float* __restrict__ w2,
                        const float* __restrict__ b2, const float* __restrict__ bias,
                        const float* __restrict__ lin, float* __restrict__ out) {
  __shared__ float red[256];
  const int b = blockIdx.x, tid = threadIdx.x;
  float s = 0.f;
  for (int n = tid; n < H1N; n += 256) {
    float v = g[n] * (H1[(size_t)b * H1N + n] - mean[n]) * istd[n] + be[n];
    v = v > 0.f ? v : 0.f;
    s += v * w2[n];
  }
  red[tid] = s;
  __syncthreads();
  for (int st = 128; st > 0; st >>= 1) {
    if (tid < st) red[tid] += red[tid + st];
    __syncthreads();
  }
  if (tid == 0) {
    float logit = lin[b] + bias[0] + b2[0] + red[0];
    out[b] = 1.0f / (1.0f + __expf(-logit));
  }
}

// ---------------- host launcher ----------------
extern "C" void kernel_launch(void* const* d_in, const int* in_sizes, int n_in,
                              void* d_out, int out_size, void* d_ws, size_t ws_size,
                              hipStream_t stream) {
  (void)in_sizes; (void)n_in; (void)out_size; (void)ws_size;

  const int*   x      = (const int*)  d_in[0];
  const float* embT   = (const float*)d_in[1];
  const float* linT   = (const float*)d_in[2];
  const float* bias   = (const float*)d_in[3];
  const float* se_w1  = (const float*)d_in[4];
  const float* se_w2  = (const float*)d_in[5];
  const float* bil_W  = (const float*)d_in[6];
  const float* w0     = (const float*)d_in[7];
  // d_in[8] = b0 : cancels under train-mode BN
  const float* g0     = (const float*)d_in[9];
  const float* be0    = (const float*)d_in[10];
  const float* w1     = (const float*)d_in[11];
  // d_in[12] = b1 : cancels under train-mode BN
  const float* g1     = (const float*)d_in[13];
  const float* be1    = (const float*)d_in[14];
  const float* w2     = (const float*)d_in[15];
  const float* b2     = (const float*)d_in[16];
  float* out = (float*)d_out;

  // workspace carve-up
  char* ws = (char*)d_ws;
  size_t off = 0;
  auto carve = [&](size_t bytes) -> void* {
    void* p = ws + off;
    off = (off + bytes + 255) & ~(size_t)255;
    return p;
  };
  u16*   Wb    = (u16*)  carve((size_t)FD * FD * 2);          // bilinear weight, bf16
  u16*   w0T   = (u16*)  carve((size_t)H0N * COMB * 2);       // w0^T bf16
  u16*   w1T   = (u16*)  carve((size_t)H1N * H0N * 2);        // w1^T bf16
  u16*   E2    = (u16*)  carve((size_t)2 * B_SZ * FD * 2);    // [emb; se_emb] bf16
  float* linb  = (float*)carve((size_t)B_SZ * 4);
  float* S     = (float*)carve((size_t)2 * B_SZ * FD * 4);    // bilinear GEMM out
  u16*   comb  = (u16*)  carve((size_t)B_SZ * COMB * 2);      // concat [p,q] bf16
  float* h0    = (float*)carve((size_t)B_SZ * H0N * 4);
  u16*   h0a   = (u16*)  carve((size_t)B_SZ * H0N * 2);
  float* h1    = (float*)carve((size_t)B_SZ * H1N * 4);
  float* m0    = (float*)carve((size_t)H0N * 4);
  float* i0    = (float*)carve((size_t)H0N * 4);
  float* m1    = (float*)carve((size_t)H1N * 4);
  float* i1    = (float*)carve((size_t)H1N * 4);

  // 1) weight repack / bf16 conversion
  k_repack_bilW<<<(FD * FD) / 256, 256, 0, stream>>>(bil_W, Wb);
  k_transpose_bf16<<<((size_t)H0N * COMB + 255) / 256, 256, 0, stream>>>(w0, w0T, COMB, H0N);
  k_transpose_bf16<<<((size_t)H1N * H0N + 255) / 256, 256, 0, stream>>>(w1, w1T, H0N, H1N);

  // 2) embedding gather + linear term + SENet
  k_embed_senet<<<B_SZ, 256, 0, stream>>>(x, embT, linT, se_w1, se_w2, E2, linb);

  // 3) bilinear GEMM: [8192 x 768] @ [768 x 768]
  k_gemm_bf16<<<dim3(FD / 128, (2 * B_SZ) / 128), 256, 0, stream>>>(E2, Wb, S,
                                                                    2 * B_SZ, FD, FD);
  // 4) diag correction + e*s + concat
  k_combine<<<B_SZ, 256, 0, stream>>>(S, E2, bil_W, comb);

  // 5) MLP layer 0 GEMM + BN stats + BN/ReLU
  k_gemm_bf16<<<dim3(H0N / 128, B_SZ / 128), 256, 0, stream>>>(comb, w0T, h0,
                                                               B_SZ, H0N, COMB);
  k_colstats<<<H0N / 64, 256, 0, stream>>>(h0, m0, i0, H0N);
  k_bnrelu_bf16<<<(B_SZ * H0N) / 256, 256, 0, stream>>>(h0, m0, i0, g0, be0, h0a,
                                                        H0N, B_SZ * H0N);
  // 6) MLP layer 1 GEMM + BN stats
  k_gemm_bf16<<<dim3(H1N / 128, B_SZ / 128), 256, 0, stream>>>(h0a, w1T, h1,
                                                               B_SZ, H1N, H0N);
  k_colstats<<<H1N / 64, 256, 0, stream>>>(h1, m1, i1, H1N);

  // 7) final fused layer
  k_final<<<B_SZ, 256, 0, stream>>>(h1, m1, i1, g1, be1, w2, b2, bias, linb, out);
}
